// UniGNN_2594160246977
// MI455X (gfx1250) — compile-verified, matrix operand order checked by reading, not run on hardware
//
#include <hip/hip_runtime.h>

// ---------- types for WMMA fragments ----------
typedef __attribute__((ext_vector_type(16))) __bf16          v16bf;
typedef __attribute__((ext_vector_type(8)))  float           v8f;
typedef __attribute__((ext_vector_type(8)))  unsigned short  u16x8;
typedef __attribute__((ext_vector_type(4)))  int             v4i;

union FragU { v16bf v; u16x8 h[2]; };

// Async global->LDS path (CDNA5 GLOBAL_LOAD_ASYNC_TO_LDS_B128, ASYNCcnt-tracked).
// Probe-confirmed signature: arg0 = v4i in AS(1) (global src), arg1 = LDS dst,
// arg2/arg3 = immediate offset / cache-policy.
#if __has_builtin(__builtin_amdgcn_global_load_async_to_lds_b128) && \
    __has_builtin(__builtin_amdgcn_s_wait_asynccnt)
#define USE_ASYNC_LDS 1
typedef __attribute__((address_space(1))) v4i* gas_v4i;
typedef __attribute__((address_space(3))) v4i* las_v4i;
#else
#define USE_ASYNC_LDS 0
#endif

// fp32 -> bf16 round-to-nearest-even
__device__ __forceinline__ unsigned short f2bf(float f) {
  unsigned u = __float_as_uint(f);
  u += 0x7FFFu + ((u >> 16) & 1u);
  return (unsigned short)(u >> 16);
}

// ---------- utility kernels ----------
__global__ void k_fill_zero(float* __restrict__ p, long long n) {
  long long i = (long long)blockIdx.x * blockDim.x + threadIdx.x;
  long long stride = (long long)gridDim.x * blockDim.x;
  for (; i < n; i += stride) p[i] = 0.0f;
}

__global__ void k_cvt_bf16(const float* __restrict__ s, unsigned short* __restrict__ d,
                           long long n) {
  long long i = (long long)blockIdx.x * blockDim.x + threadIdx.x;
  long long stride = (long long)gridDim.x * blockDim.x;
  for (; i < n; i += stride) d[i] = f2bf(s[i]);
}

// weights: [K, Cs] fp32 -> [K, Cp] bf16 (zero pad columns Cs..Cp-1)
__global__ void k_cvt_w_pad(const float* __restrict__ s, unsigned short* __restrict__ d,
                            int K, int Cs, int Cp) {
  int i = blockIdx.x * blockDim.x + threadIdx.x;
  int total = K * Cp;
  if (i >= total) return;
  int k = i / Cp, c = i % Cp;
  d[i] = (c < Cs) ? f2bf(s[k * Cs + c]) : (unsigned short)0;
}

// per-hyperedge incidence counts (float, for mean)
__global__ void k_count(const int* __restrict__ idx, float* __restrict__ cnt, int nnz) {
  int i = blockIdx.x * blockDim.x + threadIdx.x;
  if (i < nnz) atomicAdd(&cnt[idx[i]], 1.0f);
}

// ---------- bf16 WMMA GEMM:  C[M,N] = A[M,K] * B[K,N], fp32 accumulate ----------
// Block tile: 128(M) x 64(N), K stepped by 32. 8 wave32 waves; wave w owns rows
// [w*16, w*16+16) of the block and all 64 columns (4 accumulators).
__global__ __launch_bounds__(256)
void k_gemm_bf16(const unsigned short* __restrict__ A,
                 const unsigned short* __restrict__ B,
                 float* __restrict__ C, int M, int N, int K) {
  __shared__ unsigned short As[128][32];   // 8 KB
  __shared__ unsigned short Bs[64][32];    // 4 KB, stored transposed: Bs[n][k]

  const int tid  = threadIdx.x;
  const int wave = tid >> 5;
  const int lane = tid & 31;
  const int hv   = lane >> 4;     // lane half (0/1)
  const int l16  = lane & 15;
  const int m0   = blockIdx.x * 128;
  const int n0   = blockIdx.y * 64;

  v8f zero = {};
  v8f acc[4];
  acc[0] = zero; acc[1] = zero; acc[2] = zero; acc[3] = zero;

  // cooperative-load thread mappings
  const int ar   = tid >> 1;          // A row in tile (128 rows, 2 threads/row)
  const int aseg = (tid & 1) * 16;    // 16 bf16 elements each
  const int bk   = tid >> 3;          // B k-row in tile (32 rows, 8 threads/row)
  const int bn   = (tid & 7) * 8;     // 8 contiguous n elements

  // Clamp the M-boundary instead of branching: out-of-range tile rows only feed
  // waves whose stores are guarded off, so clamped (duplicate) data is discarded.
  int gr = m0 + ar;
  if (gr >= M) gr = M - 1;
  const unsigned short* agp = A + (size_t)gr * K + aseg;

  for (int k0 = 0; k0 < K; k0 += 32) {
    // --- A tile: 128x32 bf16 ---
#if USE_ASYNC_LDS
    {
      const unsigned short* gp = agp + k0;
      unsigned short* lp = &As[ar][aseg];
      __builtin_amdgcn_global_load_async_to_lds_b128((gas_v4i)gp, (las_v4i)lp, 0, 0);
      __builtin_amdgcn_global_load_async_to_lds_b128((gas_v4i)(gp + 8), (las_v4i)(lp + 8), 0, 0);
    }
#else
    {
      const u16x8* p = (const u16x8*)(agp + k0);
      u16x8 v0 = p[0];
      u16x8 v1 = p[1];
      *(u16x8*)&As[ar][aseg]     = v0;
      *(u16x8*)&As[ar][aseg + 8] = v1;
    }
#endif
    // --- B tile: read row-major coalesced, transpose into Bs[n][k] ---
    {
      const u16x8* p = (const u16x8*)(B + (size_t)(k0 + bk) * N + n0 + bn);
      u16x8 v = p[0];
      #pragma unroll
      for (int j = 0; j < 8; ++j) Bs[bn + j][bk] = v[j];
      if (k0 + 32 < K)  // hint next B slab into cache (global_prefetch_b8)
        __builtin_prefetch(B + (size_t)(k0 + 32 + bk) * N + n0 + bn, 0, 1);
    }
#if USE_ASYNC_LDS
    __builtin_amdgcn_s_wait_asynccnt(0);
#endif
    __syncthreads();

    // --- fragment assembly per documented 16-bit 16x32 layout:
    //     lane half 0 -> K {0..7,16..23}, half 1 -> K {8..15,24..31} ---
    const int koff = hv * 8;
    FragU fa;
    fa.h[0] = *(const u16x8*)&As[wave * 16 + l16][koff];
    fa.h[1] = *(const u16x8*)&As[wave * 16 + l16][16 + koff];

    #pragma unroll
    for (int nt = 0; nt < 4; ++nt) {
      FragU fb;
      fb.h[0] = *(const u16x8*)&Bs[nt * 16 + l16][koff];
      fb.h[1] = *(const u16x8*)&Bs[nt * 16 + l16][16 + koff];
      acc[nt] = __builtin_amdgcn_wmma_f32_16x16x32_bf16(
          false, fa.v, false, fb.v, (short)0, acc[nt], false, false);
    }
    __syncthreads();
  }

  // --- store per 16x16 f32 C/D layout: VGPR r -> M = r + 8*half, N = lane%16 ---
  const int mrow = m0 + wave * 16;
  if (mrow < M) {
    #pragma unroll
    for (int nt = 0; nt < 4; ++nt) {
      const int col = n0 + nt * 16 + l16;
      #pragma unroll
      for (int r = 0; r < 8; ++r) {
        const int row = mrow + r + 8 * hv;
        C[(size_t)row * N + col] = acc[nt][r];
      }
    }
  }
}

// ---------- scatter-add: dst[didx[i]] += src[sidx[i]], full rows of C floats ----------
__global__ void k_scatter_add(const float* __restrict__ src, const int* __restrict__ sidx,
                              const int* __restrict__ didx, float* __restrict__ dst, int C) {
  const int i = blockIdx.x;
  const float* s = src + (size_t)sidx[i] * C;
  float*       d = dst + (size_t)didx[i] * C;
  for (int c = threadIdx.x; c < C; c += blockDim.x)
    atomicAdd(&d[c], s[c]);
}

// Xe[e] *= degE[e] / max(cnt[e], 1)
__global__ void k_scale_edges(float* __restrict__ Xe, const float* __restrict__ degE,
                              const float* __restrict__ cnt, int C) {
  const int e = blockIdx.x;
  const float scl = degE[e] / fmaxf(cnt[e], 1.0f);
  float* row = Xe + (size_t)e * C;
  for (int c = threadIdx.x; c < C; c += blockDim.x) row[c] *= scl;
}

// Xv[v] *= degV[v], optional ReLU
__global__ void k_scale_nodes(float* __restrict__ Xv, const float* __restrict__ degV,
                              int C, int do_relu) {
  const int v = blockIdx.x;
  const float scl = degV[v];
  float* row = Xv + (size_t)v * C;
  for (int c = threadIdx.x; c < C; c += blockDim.x) {
    float x = row[c] * scl;
    if (do_relu) x = fmaxf(x, 0.0f);
    row[c] = x;
  }
}

// log_softmax over first C of ld padded columns, one thread per row
__global__ void k_logsoftmax(const float* __restrict__ L, float* __restrict__ out,
                             int N, int ld, int C) {
  const int r = blockIdx.x * blockDim.x + threadIdx.x;
  if (r >= N) return;
  const float* x = L + (size_t)r * ld;
  float mx = -3.0e38f;
  for (int c = 0; c < C; ++c) mx = fmaxf(mx, x[c]);
  float s = 0.0f;
  for (int c = 0; c < C; ++c) s += __expf(x[c] - mx);
  const float lg = __logf(s);
  for (int c = 0; c < C; ++c) out[(size_t)r * C + c] = x[c] - mx - lg;
}

// ---------- host launcher ----------
extern "C" void kernel_launch(void* const* d_in, const int* in_sizes, int n_in,
                              void* d_out, int out_size, void* d_ws, size_t ws_size,
                              hipStream_t stream) {
  (void)n_in; (void)out_size; (void)ws_size;
  const float* X      = (const float*)d_in[0];
  const int*   vertex = (const int*)  d_in[1];
  const int*   edges  = (const int*)  d_in[2];
  const float* W1     = (const float*)d_in[3];
  const float* W2     = (const float*)d_in[4];
  const float* Wout   = (const float*)d_in[5];
  const float* degE   = (const float*)d_in[6];
  const float* degV   = (const float*)d_in[7];
  float* out = (float*)d_out;

  const int Nn = 50000, Me = 25000, C = 512, CLP = 64, CL = 40;
  const int NNZv = in_sizes[1];

  // workspace carve-out (256B aligned)
  char* w = (char*)d_ws;
  auto carve = [&](size_t bytes) {
    char* p = w;
    w += (bytes + 255) & ~(size_t)255;
    return p;
  };
  unsigned short* Xb = (unsigned short*)carve((size_t)Nn * C * 2);   // bf16 activations
  unsigned short* Wb = (unsigned short*)carve((size_t)C * C * 2);    // bf16 weights (padded)
  float* Y   = (float*)carve((size_t)Nn * C * 4);                    // GEMM output
  float* Xe  = (float*)carve((size_t)Me * C * 4);                    // edge accumulators
  float* H   = (float*)carve((size_t)Nn * C * 4);                    // hidden activations
  float* LOG = (float*)carve((size_t)Nn * CLP * 4);                  // padded logits
  float* cnt = (float*)carve((size_t)Me * 4);                        // edge cardinalities

  // per-edge counts (recomputed every call; no persistent state)
  k_fill_zero<<<256, 256, 0, stream>>>(cnt, (long long)Me);
  k_count<<<(NNZv + 255) / 256, 256, 0, stream>>>(edges, cnt, NNZv);

  auto conv = [&](const float* Xin, const float* Wf, int Cout_src, int Cout_pad,
                  float* Xout, int do_relu) {
    // 1) bf16 convert (X fully consumed here -> Xout may alias Xin)
    k_cvt_bf16<<<2048, 256, 0, stream>>>(Xin, Xb, (long long)Nn * C);
    const int wtot = C * Cout_pad;
    k_cvt_w_pad<<<(wtot + 255) / 256, 256, 0, stream>>>(Wf, Wb, C, Cout_src, Cout_pad);
    // 2) WMMA GEMM
    dim3 gg((Nn + 127) / 128, Cout_pad / 64);
    k_gemm_bf16<<<gg, 256, 0, stream>>>(Xb, Wb, Y, Nn, Cout_pad, C);
    // 3) vertex -> edge mean * degE
    const int tb = (Cout_pad >= 256) ? 256 : Cout_pad;
    k_fill_zero<<<1024, 256, 0, stream>>>(Xe, (long long)Me * Cout_pad);
    k_scatter_add<<<NNZv, tb, 0, stream>>>(Y, vertex, edges, Xe, Cout_pad);
    k_scale_edges<<<Me, tb, 0, stream>>>(Xe, degE, cnt, Cout_pad);
    // 4) edge -> vertex sum * degV (+ReLU)
    k_fill_zero<<<1024, 256, 0, stream>>>(Xout, (long long)Nn * Cout_pad);
    k_scatter_add<<<NNZv, tb, 0, stream>>>(Xe, edges, vertex, Xout, Cout_pad);
    k_scale_nodes<<<Nn, tb, 0, stream>>>(Xout, degV, Cout_pad, do_relu);
  };

  conv(X, W1, 512, 512, H, 1);     // layer 1: d_in -> H
  conv(H, W2, 512, 512, H, 1);     // layer 2: H -> H (safe: H consumed at cvt)
  conv(H, Wout, 40, 64, LOG, 0);   // layer 3: H -> padded logits

  k_logsoftmax<<<(Nn + 255) / 256, 256, 0, stream>>>(LOG, out, Nn, CLP, CL);
}
